// PCILTConv2d_70892730187911
// MI455X (gfx1250) — 8-bit matrix operand order repaired and checked by reading, compiled, not run on hardware
//
#include <hip/hip_runtime.h>


typedef int v8i __attribute__((ext_vector_type(8)));
typedef int v4i __attribute__((ext_vector_type(4)));

#define KW 576


__global__ void k_build_limbs(const float* __restrict__ w,
                              signed char* __restrict__ limb) {
    __shared__ int qw[KW];
    const int o = blockIdx.x;
    const int i = threadIdx.x;
    for (int j = threadIdx.x; j < KW; j += 256) {
        int q = (int)rintf(w[o * KW + j] * 255.0f);
        q = q < 0 ? 0 : (q > 255 ? 255 : q);
        qw[j] = q;
    }
    __syncthreads();
    int s = 0;
    for (int j = 0; j < KW; ++j) {
        int p = qw[j] * i;
        s += ((p & 0xFFFF) ^ 0x8000) - 0x8000;
    }
    int r = s;
#pragma unroll
    for (int L = 0; L < 4; ++L) {
        signed char l = (signed char)(r & 0xFF);
        limb[L * 32768 + i * 128 + o] = l;
        r = (r - (int)l) >> 8;
    }
}

__global__ void k_build_bswz(const signed char* __restrict__ limb,
                             unsigned int* __restrict__ bswz) {
    const int d    = blockIdx.x * 256 + threadIdx.x;
    const int v    = d & 7;
    const int lane = (d >> 3) & 31;
    const int s    = (d >> 8) & 3;
    const int L    = (d >> 10) & 3;
    const int ot   = d >> 12;
    const int half = lane >> 4;
    const int n    = lane & 15;
    unsigned int word = 0;
#pragma unroll
    for (int b2 = 0; b2 < 4; ++b2) {
        int K = (v >= 4 ? 32 : 0) + half * 16 + (v & 3) * 4 + b2;
        int i = s * 64 + K;
        unsigned int by =
            (unsigned int)(unsigned char)limb[L * 32768 + i * 128 + (ot * 16 + n)];
        word |= by << (b2 * 8);
    }
    bswz[d] = word;
}

__launch_bounds__(128, 1)
__global__ void k_main(const float* __restrict__ x,
                       const float* __restrict__ bias,
                       const unsigned int* __restrict__ bswz,
                       float* __restrict__ out) {
    extern __shared__ char smem[];
    int*          s_b = (int*)smem;
    unsigned int* s_h = (unsigned int*)(smem + 131072);
    int*          s_a = (int*)(smem + 131072 + 73728);

    const int tid  = threadIdx.x;
    const int lane = tid & 31;
    const int wave = tid >> 5;
    const int x0 = blockIdx.x * 32;
    const int y0 = blockIdx.y * 4;
    const int b  = blockIdx.z;

    for (int idx = tid; idx < 8192; idx += 128) {
        unsigned int ldsa = (unsigned int)(uintptr_t)(s_b + idx * 4);
        unsigned long long ga = (unsigned long long)(uintptr_t)(bswz + idx * 4);
        asm volatile("global_load_async_to_lds_b128 %0, %1, off"
                     :: "v"(ldsa), "v"(ga) : "memory");
    }
    for (int idx = tid; idx < 18432; idx += 128) s_h[idx] = 0u;
    asm volatile("s_wait_asynccnt 0x0" ::: "memory");
    __syncthreads();

    {
        const int pix  = lane & 15;
        const int half = lane >> 4;
        for (int pt = wave; pt < 18; pt += 4) {
            const int row = pt / 3;
            const int ci  = (pt % 3) * 16 + pix;
            const int yg  = y0 - 1 + row;
            const int xg  = x0 - 8 + ci;
            if (yg >= 0 && yg < 128 && xg >= 0 && xg < 128) {
                const float* px = x + ((size_t)b * 64 + half * 32) * 16384
                                    + (size_t)yg * 128 + xg;
                unsigned int* hrow = s_h + (pt * 16 + pix) * 64;
                for (int cc = 0; cc < 32; ++cc) {
                    float xv = px[(size_t)cc * 16384];
                    int q = (int)rintf(xv * 255.0f);
                    q = q < 0 ? 0 : (q > 255 ? 255 : q);
                    int g  = q >> 2;
                    int pg = (g & 51) | ((g & 4) << 1) | ((g & 8) >> 1);
                    atomicAdd(&hrow[pg], 1u << ((q & 3) * 8));
                }
            }
        }
    }
    __syncthreads();

    const v8i vzero = {0, 0, 0, 0, 0, 0, 0, 0};
    for (int ot = 0; ot < 8; ++ot) {
        const int pix  = lane & 15;
        const int half = lane >> 4;

        v8i Bv[16];
#pragma unroll
        for (int L = 0; L < 4; ++L) {
#pragma unroll
            for (int s = 0; s < 4; ++s) {
                const v4i* bb = (const v4i*)(const void*)
                    (s_b + ((((ot * 4 + L) * 4 + s) * 32 + lane) * 8));
                v4i lo = bb[0], hi = bb[1];
                Bv[L * 4 + s] = __builtin_shufflevector(lo, hi, 0,1,2,3,4,5,6,7);
            }
        }

        for (int pt = wave; pt < 18; pt += 4) {
            const v4i* hb = (const v4i*)(const void*)s_h
                          + ((pt * 16 + pix) * 16 + half * 2);
            v8i a[4];
#pragma unroll
            for (int s = 0; s < 4; ++s) {
                v4i lo = hb[s * 4 + 0], hi = hb[s * 4 + 1];
                a[s] = __builtin_shufflevector(lo, hi, 0,1,2,3,4,5,6,7);
            }

            v8i acc[16];
#pragma unroll
            for (int s = 0; s < 4; ++s) {
#pragma unroll
                for (int L = 0; L < 4; ++L) {
                    acc[s * 4 + L] = __builtin_amdgcn_wmma_i32_16x16x64_iu8(
                        false, a[s], true, Bv[L * 4 + s], vzero, false, false);
                }
            }

            const int row = pt / 3;
            const int cib = (pt % 3) * 16;
#pragma unroll
            for (int k = 0; k < 8; ++k) {
                unsigned int c0 = (unsigned int)acc[0][k]  + (unsigned int)acc[4][k]
                                + (unsigned int)acc[8][k]  + (unsigned int)acc[12][k];
                unsigned int c1 = (unsigned int)acc[1][k]  + (unsigned int)acc[5][k]
                                + (unsigned int)acc[9][k]  + (unsigned int)acc[13][k];
                unsigned int c2 = (unsigned int)acc[2][k]  + (unsigned int)acc[6][k]
                                + (unsigned int)acc[10][k] + (unsigned int)acc[14][k];
                unsigned int c3 = (unsigned int)acc[3][k]  + (unsigned int)acc[7][k]
                                + (unsigned int)acc[11][k] + (unsigned int)acc[15][k];
                unsigned int val = c0 + (c1 << 8) + (c2 << 16) + (c3 << 24);
                int M = half * 8 + k;
                s_a[(row * 48 + cib + M) * 16 + (lane & 15)] = (int)val;
            }
        }
        __syncthreads();

        {
            const int n  = tid & 15;
            const int c8 = tid >> 4;
            const float bv = bias[ot * 16 + n];
            for (int j = 0; j < 4; ++j) {
                const int xl = c8 + j * 8;
                int rs[6];
#pragma unroll
                for (int r = 0; r < 6; ++r) {
                    rs[r] = s_a[(r * 48 + xl + 7) * 16 + n]
                          + s_a[(r * 48 + xl + 8) * 16 + n]
                          + s_a[(r * 48 + xl + 9) * 16 + n];
                }
#pragma unroll
                for (int ry = 0; ry < 4; ++ry) {
                    int acc2 = rs[ry] + rs[ry + 1] + rs[ry + 2];
                    const int y  = y0 + ry;
                    const int xx = x0 + xl;
                    out[(((size_t)b * 128 + ot * 16 + n) << 14) + (y << 7) + xx] =
                        (float)acc2 + bv;
                }
            }
        }
        __syncthreads();
    }
}

extern "C" void kernel_launch(void* const* d_in, const int* in_sizes, int n_in,
                              void* d_out, int out_size, void* d_ws, size_t ws_size,
                              hipStream_t stream) {
    (void)in_sizes; (void)n_in; (void)out_size; (void)ws_size;
    const float* x    = (const float*)d_in[0];
    const float* w    = (const float*)d_in[1];
    const float* bias = (const float*)d_in[2];
    float* out = (float*)d_out;

    signed char*  limb = (signed char*)d_ws;
    unsigned int* bswz = (unsigned int*)((char*)d_ws + 131072);

    k_build_limbs<<<128, 256, 0, stream>>>(w, limb);
    k_build_bswz<<<128, 256, 0, stream>>>(limb, bswz);
    k_main<<<dim3(4, 32, 8), 128, 223232, stream>>>(x, bias, bswz, out);
}
